// Net_6605659702095
// MI455X (gfx1250) — matrix operand order repaired and checked by reading, not compiled, not run on hardware
//
#include <hip/hip_runtime.h>

typedef __attribute__((ext_vector_type(16))) _Float16 v16h;
typedef __attribute__((ext_vector_type(8)))  float    v8f;

#define N_NODES 4096
#define N_EDGES 131072
#define HID     8
#define M_DIM   (N_NODES * HID)   // 32768 = GEMV K dimension
#define Y_DIM   4096
#define KSPLIT  64
#define KCHUNK  (M_DIM / KSPLIT)  // 512, multiple of 32

// ---------------- front-end (GCN conv collapses to scalar aggregation) ----------

__global__ void init_kernel(float* deg, float* agg, float* y, const float* br) {
  int i = blockIdx.x * blockDim.x + threadIdx.x;
  if (i < N_NODES) {
    deg[i] = 1.0f;   // self loop
    agg[i] = 0.0f;
    y[i]   = br[i];  // bias init; GEMV atomically accumulates on top
  }
}

__global__ void degree_kernel(const int* __restrict__ ei, float* __restrict__ deg) {
  int e = blockIdx.x * blockDim.x + threadIdx.x;
  if (e < N_EDGES) {
    int d = ei[N_EDGES + e];           // edge_index[1][e]
    atomicAdd(&deg[d], 1.0f);
  }
}

__global__ void dinv_kernel(const float* __restrict__ deg, float* __restrict__ dinv) {
  int i = blockIdx.x * blockDim.x + threadIdx.x;
  if (i < N_NODES) dinv[i] = rsqrtf(deg[i]);   // deg >= 1 always
}

__global__ void scatter_kernel(const int* __restrict__ ei, const float* __restrict__ x,
                               const float* __restrict__ dinv, float* __restrict__ agg) {
  int e = blockIdx.x * blockDim.x + threadIdx.x;
  if (e < N_EDGES) {
    int s = ei[e];
    int d = ei[N_EDGES + e];
    atomicAdd(&agg[d], x[s] * dinv[s] * dinv[d]);
  }
}

__global__ void h_kernel(const float* __restrict__ x, const float* __restrict__ agg,
                         const float* __restrict__ dinv, const float* __restrict__ W1,
                         const float* __restrict__ b1, _Float16* __restrict__ h) {
  int idx = blockIdx.x * blockDim.x + threadIdx.x;
  if (idx < M_DIM) {
    int i = idx >> 3, j = idx & 7;
    float total = agg[i] + x[i] * dinv[i] * dinv[i];   // add self-loop message
    float v = total * W1[j] + b1[j];
    h[idx] = (_Float16)fmaxf(v, 0.0f);                 // ReLU
  }
}

// ---------------- bandwidth-bound GEMV via WMMA ---------------------------------
// y[col] += sum_k h[k] * Wr[k][col].  A fragment: h in row M=0, rows 1..15 zero.
// A layout (16-bit 16x32, contiguous k order chosen here): lane 0 -> M=0,K={0..15};
//                                      lane 16 -> M=0,K={16..31}.
// B layout (32x16): lane L, half i -> K = 16*(L>=16) + i, N = L & 15.
// D row M=0 lives in acc[0] on lanes 0..15 (N = lane).

__global__ void __launch_bounds__(256)
gemv_wmma(const _Float16* __restrict__ h, const float* __restrict__ Wr,
          float* __restrict__ y) {
  const int lane    = threadIdx.x & 31;
  const int wave    = threadIdx.x >> 5;
  const int colbase = blockIdx.x * 128 + wave * 16;
  const int col     = colbase + (lane & 15);
  const int kb      = blockIdx.y * KCHUNK;

  const bool a_active = (lane == 0) || (lane == 16);
  const int  a_off    = (lane == 16) ? 16 : 0;
  const int  b_krow   = (lane >> 4) << 4;   // 0 for lanes 0-15, 16 for lanes 16-31

  v8f acc = {};

  for (int k0 = kb; k0 < kb + KCHUNK; k0 += 32) {
    // A fragment: only M=0 row nonzero
    v16h a;
#pragma unroll
    for (int t = 0; t < 16; ++t) {
      int k = k0 + a_off + t;
      a[t] = a_active ? h[k] : (_Float16)0.0f;
    }
    // B fragment: 32x16 tile of Wr, f32 -> f16
    v16h b;
    const float* wp = Wr + (size_t)(k0 + b_krow) * Y_DIM + col;
#pragma unroll
    for (int i = 0; i < 16; ++i) {
      b[i] = (_Float16)wp[(size_t)i * Y_DIM];
    }
    acc = __builtin_amdgcn_wmma_f32_16x16x32_f16(
        /*neg_a=*/false, a, /*neg_b=*/false, b,
        /*c_mod=*/(short)0, acc, /*reuse_a=*/false, /*reuse_b=*/false);
  }

  if (lane < 16) atomicAdd(&y[colbase + lane], acc[0]);
}

// ---------------- launch ---------------------------------------------------------

extern "C" void kernel_launch(void* const* d_in, const int* in_sizes, int n_in,
                              void* d_out, int out_size, void* d_ws, size_t ws_size,
                              hipStream_t stream) {
  const float* x  = (const float*)d_in[0];
  const int*   ei = (const int*)d_in[1];    // edge_index [2, E] (int32 under JAX default x64-off)
  const float* W1 = (const float*)d_in[2];
  const float* b1 = (const float*)d_in[3];
  const float* Wr = (const float*)d_in[4];
  const float* br = (const float*)d_in[5];
  float*       y  = (float*)d_out;

  float*    deg  = (float*)d_ws;
  float*    dinv = deg + N_NODES;
  float*    agg  = dinv + N_NODES;
  _Float16* hbuf = (_Float16*)(agg + N_NODES);   // 32768 halves

  init_kernel  <<<(N_NODES + 255) / 256, 256, 0, stream>>>(deg, agg, y, br);
  degree_kernel<<<(N_EDGES + 255) / 256, 256, 0, stream>>>(ei, deg);
  dinv_kernel  <<<(N_NODES + 255) / 256, 256, 0, stream>>>(deg, dinv);
  scatter_kernel<<<(N_EDGES + 255) / 256, 256, 0, stream>>>(ei, x, dinv, agg);
  h_kernel     <<<(M_DIM + 255) / 256, 256, 0, stream>>>(x, agg, dinv, W1, b1, hbuf);
  gemv_wmma    <<<dim3(32, KSPLIT), 256, 0, stream>>>(hbuf, Wr, y);
}
